// TemporalAttention_21543555957159
// MI455X (gfx1250) — matrix operand order repaired and checked by reading, not compiled, not run on hardware
//
#include <hip/hip_runtime.h>
#include <hip/hip_bf16.h>

// ---------------------------------------------------------------------------
// TemporalAttention for MI455X (gfx1250): bf16 WMMA everywhere, fp32 accum.
// One block per (batch, spatial) attention problem: grid = B*S = 1024.
// ---------------------------------------------------------------------------

typedef __attribute__((ext_vector_type(16))) __bf16 bf16x16;
typedef __attribute__((ext_vector_type(8)))  float  f32x8;
typedef __attribute__((ext_vector_type(4)))  float  f32x4;   // POD 16B vector

union Frag { bf16x16 v; f32x4 f4[2]; };
union Acc  { f32x8  v; float f[8]; };

// ---- problem constants ----
#define NBATCH 4
#define SDIM   256
#define DMODEL 512
#define TTOT   32          // 8 vp + 8 vf + 8 ap + 8 af tokens
#define NHEAD  8
#define HD     64
#define VPSZ   (4*8*256*512)   // elements of v_p_out (== v_f_out)

// ---- LDS layout (byte offsets into dynamic shared) ----
#define XO_STR  520                      // bf16 elems per row (pad for banks/align)
#define XO_OFF  0                        // x (phase 1) then O (phases 4-5): 32x520 bf16
#define Q_OFF   33280                    // q bf16 32x520
#define K_OFF   66560                    // k bf16 32x520
#define VT_OFF  99840                    // v^T bf16 512x40
#define VT_STR  40
#define S_OFF   140800                   // scores f32 8*32*32
#define P_OFF   173568                   // probs bf16 8*32*40
#define P_STR   40
#define SMEM_BYTES 194048

#define QKV_PACK_N  (96*16*32*16)        // bf16 elems of packed W_qkv
#define PROJ_PACK_N (32*16*32*16)

__device__ inline f32x8 wmma_bf16(bf16x16 a, bf16x16 b, f32x8 c) {
  return __builtin_amdgcn_wmma_f32_16x16x32_bf16(
      /*neg_a=*/false, a, /*neg_b=*/false, b,
      /*c_mod=*/(short)0, c, /*reuse_a=*/false, /*reuse_b=*/false);
}

// A-fragment (16x32 bf16): lane reads row (lane&15+16*mt), two 16B chunks at
// k = kbase and k = kbase+16  (kbase = kc*32 + 8*(lane>>4))
__device__ inline bf16x16 ldsA(const __bf16* base, int row, int rowstr, int kbase) {
  Frag u;
  const __bf16* p = base + row * rowstr + kbase;
  u.f4[0] = *(const f32x4*)(p);
  u.f4[1] = *(const f32x4*)(p + 16);
  return u.v;
}

// B-fragment stored contiguously (32 bytes per lane)
__device__ inline bf16x16 ldB16(const __bf16* p) {
  Frag u;
  u.f4[0] = ((const f32x4*)p)[0];
  u.f4[1] = ((const f32x4*)p)[1];
  return u.v;
}

// ---------------------------------------------------------------------------
// Pre-pack W_qkv (512x1536) and W_proj (512x512) f32 -> bf16 in exact WMMA
// B-fragment order: dst[((nt*16+kc)*32+lane)*16 + i] = W[kc*32+8*(lane>>4)+(i&7)+16*(i>>3)][nt*16+(lane&15)]
// ---------------------------------------------------------------------------
__global__ void pack_weights_kernel(const float* __restrict__ Wqkv,
                                    const float* __restrict__ Wproj,
                                    __bf16* __restrict__ ws) {
  for (int e = blockIdx.x * blockDim.x + threadIdx.x;
       e < QKV_PACK_N + PROJ_PACK_N; e += gridDim.x * blockDim.x) {
    if (e < QKV_PACK_N) {
      int i = e & 15, lane = (e >> 4) & 31, kc = (e >> 9) & 15, nt = e >> 13;
      int k = kc * 32 + ((lane >> 4) << 3) + (i & 7) + ((i >> 3) << 4);
      int n = nt * 16 + (lane & 15);
      ws[e] = (__bf16)Wqkv[k * 1536 + n];
    } else {
      int e2 = e - QKV_PACK_N;
      int i = e2 & 15, lane = (e2 >> 4) & 31, kc = (e2 >> 9) & 15, nt = e2 >> 13;
      int k = kc * 32 + ((lane >> 4) << 3) + (i & 7) + ((i >> 3) << 4);
      int n = nt * 16 + (lane & 15);
      ws[e] = (__bf16)Wproj[k * 512 + n];
    }
  }
}

// Zero the a_p_out / a_f_out tail (accumulated via atomics) each call.
__global__ void zero_tail_kernel(float* __restrict__ out) {
  int i = blockIdx.x * blockDim.x + threadIdx.x;   // 128*256 = 32768 elems
  out[2 * VPSZ + i] = 0.0f;
}

// ---------------------------------------------------------------------------
// Main fused attention kernel
// ---------------------------------------------------------------------------
__global__ __launch_bounds__(256)
void attn_kernel(const float* __restrict__ vp, const float* __restrict__ vf,
                 const float* __restrict__ ap, const float* __restrict__ af,
                 const float* __restrict__ bproj,
                 const __bf16* __restrict__ wpack,
                 float* __restrict__ out) {
  extern __shared__ __align__(16) char smem[];
  __bf16* xO = (__bf16*)(smem + XO_OFF);
  __bf16* qb = (__bf16*)(smem + Q_OFF);
  __bf16* kb = (__bf16*)(smem + K_OFF);
  __bf16* vT = (__bf16*)(smem + VT_OFF);
  float*  Sb = (float*) (smem + S_OFF);
  __bf16* Pb = (__bf16*)(smem + P_OFF);

  const int tid  = threadIdx.x;
  const int wave = tid >> 5;
  const int lane = tid & 31;
  const int lo   = lane & 15;
  const int hi   = lane >> 4;
  const int bs   = blockIdx.x;
  const int b    = bs >> 8;          // batch
  const int s    = bs & 255;         // spatial

  // ---- Phase 0: gather x (32 tokens x 512) -> LDS bf16 -------------------
  for (int idx = tid; idx < TTOT * 128; idx += 256) {
    int r  = idx >> 7;
    int c4 = (idx & 127) << 2;
    const float* src;
    if (r < 8)        src = vp + (((b * 8 + r)       * 256 + s) << 9) + c4;
    else if (r < 16)  src = vf + (((b * 8 + (r - 8)) * 256 + s) << 9) + c4;
    else if (r < 24)  src = ap + ((b * 8 + (r - 16)) << 9) + c4;
    else              src = af + ((b * 8 + (r - 24)) << 9) + c4;
    f32x4 v = *(const f32x4*)src;
    __bf16* d = xO + r * XO_STR + c4;
    d[0] = (__bf16)v.x; d[1] = (__bf16)v.y; d[2] = (__bf16)v.z; d[3] = (__bf16)v.w;
  }
  __syncthreads();

  // ---- Phase 1: qkv = x @ W_qkv ; RoPE fused on q,k ; v stored transposed -
  // units 0..63 : (mt, q|k, head, sub) column-pair tiles (nt, nt+2) so the
  //               RoPE pair (j, j+32) sits in this lane's two accumulators.
  // units 64..127: v tiles, stored as vT[dim][token].
  for (int u = wave; u < 128; u += 8) {
    if (u < 64) {
      int mt = u & 1, t2 = u >> 1, qk = t2 & 1, pr = t2 >> 1;
      int h = pr >> 1, sub = pr & 1;
      int nt0 = qk * 32 + h * 4 + sub;
      int nt1 = nt0 + 2;
      Acc a0 = {}, a1 = {};
      for (int kc = 0; kc < 16; ++kc) {
        bf16x16 A  = ldsA(xO, mt * 16 + lo, XO_STR, kc * 32 + 8 * hi);
        bf16x16 B0 = ldB16(wpack + (((nt0 * 16 + kc) * 32 + lane) << 4));
        bf16x16 B1 = ldB16(wpack + (((nt1 * 16 + kc) * 32 + lane) << 4));
        a0.v = wmma_bf16(A, B0, a0.v);
        a1.v = wmma_bf16(A, B1, a1.v);
      }
      int j = sub * 16 + lo;                           // rotary dim in [0,32)
      float invf   = __powf(10000.0f, -(float)j * (1.0f / 32.0f));
      float qscale = (qk == 0) ? 0.125f : 1.0f;        // hd^-0.5 folded into q
      __bf16* dst = (qk == 0) ? qb : kb;
      int col0 = h * 64 + sub * 16 + lo;
      for (int r = 0; r < 8; ++r) {
        int t = mt * 16 + r + 8 * hi;                  // token position
        float ang = (float)t * invf;
        float sn, cs; __sincosf(ang, &sn, &cs);
        float x1 = a0.f[r], x2 = a1.f[r];
        dst[t * XO_STR + col0]      = (__bf16)((x1 * cs - x2 * sn) * qscale);
        dst[t * XO_STR + col0 + 32] = (__bf16)((x2 * cs + x1 * sn) * qscale);
      }
    } else {
      int vu = u - 64, mt = vu & 1, ntv = 64 + (vu >> 1);
      Acc a0 = {};
      for (int kc = 0; kc < 16; ++kc) {
        bf16x16 A = ldsA(xO, mt * 16 + lo, XO_STR, kc * 32 + 8 * hi);
        bf16x16 B = ldB16(wpack + (((ntv * 16 + kc) * 32 + lane) << 4));
        a0.v = wmma_bf16(A, B, a0.v);
      }
      int dim = (ntv - 64) * 16 + lo;
      for (int r = 0; r < 8; ++r)
        vT[dim * VT_STR + mt * 16 + 8 * hi + r] = (__bf16)a0.f[r];
    }
  }
  __syncthreads();

  // ---- Phase 2: S = q @ k^T per head (scale already in q) ----------------
  for (int u = wave; u < 32; u += 8) {
    int h = u >> 2, mt = (u >> 1) & 1, ntk = u & 1;
    Acc a0 = {};
    for (int kc2 = 0; kc2 < 2; ++kc2) {
      bf16x16 A = ldsA(qb, mt * 16 + lo, XO_STR, h * 64 + kc2 * 32 + 8 * hi);
      bf16x16 B = ldsA(kb, ntk * 16 + lo, XO_STR, h * 64 + kc2 * 32 + 8 * hi);
      a0.v = wmma_bf16(A, B, a0.v);
    }
    for (int r = 0; r < 8; ++r)
      Sb[(h * 32 + mt * 16 + r + 8 * hi) * 32 + ntk * 16 + lo] = a0.f[r];
  }
  __syncthreads();

  // ---- Phase 3: softmax, one thread per (head, row) = 256 rows -----------
  {
    int h = tid >> 5, m = tid & 31;
    const float* srow = Sb + (h * 32 + m) * 32;
    float mx = srow[0];
    for (int n = 1; n < 32; ++n) mx = fmaxf(mx, srow[n]);
    float e[32], sum = 0.0f;
    for (int n = 0; n < 32; ++n) { e[n] = __expf(srow[n] - mx); sum += e[n]; }
    float inv = 1.0f / sum;
    __bf16* prow = Pb + (h * 32 + m) * P_STR;
    for (int n = 0; n < 32; ++n) prow[n] = (__bf16)(e[n] * inv);
  }
  __syncthreads();

  // ---- Phase 4: O = P @ V (K = 32 tokens -> one WMMA per tile) -----------
  for (int u = wave; u < 64; u += 8) {
    int h = u >> 3, rem = u & 7, mt = rem >> 2, nd = rem & 3;
    bf16x16 A = ldsA(Pb, h * 32 + mt * 16 + lo, P_STR, 8 * hi);
    bf16x16 B = ldsA(vT, h * 64 + nd * 16 + lo, VT_STR, 8 * hi);
    Acc a0 = {};
    a0.v = wmma_bf16(A, B, a0.v);
    int col = h * 64 + nd * 16 + lo;
    for (int r = 0; r < 8; ++r)
      xO[(mt * 16 + r + 8 * hi) * XO_STR + col] = (__bf16)a0.f[r];
  }
  __syncthreads();

  // ---- Phase 5: out = O @ W_proj + b ; scatter to 4 output regions -------
  const __bf16* wp = wpack + QKV_PACK_N;
  for (int u = wave; u < 64; u += 8) {
    int mt = u & 1, nt = u >> 1;
    Acc a0 = {};
    for (int kc = 0; kc < 16; ++kc) {
      bf16x16 A = ldsA(xO, mt * 16 + lo, XO_STR, kc * 32 + 8 * hi);
      bf16x16 B = ldB16(wp + (((nt * 16 + kc) * 32 + lane) << 4));
      a0.v = wmma_bf16(A, B, a0.v);
    }
    int col = nt * 16 + lo;
    float bias = bproj[col];
    for (int r = 0; r < 8; ++r) {
      float val = a0.f[r] + bias;
      int t = mt * 16 + r + 8 * hi;
      if (t < 8)
        out[(((b * 8 + t) * 256 + s) << 9) + col] = val;
      else if (t < 16)
        out[VPSZ + (((b * 8 + (t - 8)) * 256 + s) << 9) + col] = val;
      else if (t < 24)
        atomicAdd(out + 2 * VPSZ + ((b * 8 + (t - 16)) << 9) + col,
                  val * (1.0f / 256.0f));
      else
        atomicAdd(out + 2 * VPSZ + 16384 + ((b * 8 + (t - 24)) << 9) + col,
                  val * (1.0f / 256.0f));
    }
  }
}

// ---------------------------------------------------------------------------
extern "C" void kernel_launch(void* const* d_in, const int* in_sizes, int n_in,
                              void* d_out, int out_size, void* d_ws, size_t ws_size,
                              hipStream_t stream) {
  const float* vp    = (const float*)d_in[0];
  const float* vf    = (const float*)d_in[1];
  const float* ap    = (const float*)d_in[2];
  const float* af    = (const float*)d_in[3];
  const float* Wqkv  = (const float*)d_in[4];
  const float* Wproj = (const float*)d_in[5];
  const float* bproj = (const float*)d_in[6];
  float*  out = (float*)d_out;
  __bf16* ws  = (__bf16*)d_ws;

  hipFuncSetAttribute(reinterpret_cast<const void*>(attn_kernel),
                      hipFuncAttributeMaxDynamicSharedMemorySize, SMEM_BYTES);

  pack_weights_kernel<<<512, 256, 0, stream>>>(Wqkv, Wproj, ws);
  zero_tail_kernel<<<128, 256, 0, stream>>>(out);
  attn_kernel<<<NBATCH * SDIM, 256, SMEM_BYTES, stream>>>(
      vp, vf, ap, af, bproj, ws, out);
}
